// RoIProcessingNetwork_63032940036238
// MI455X (gfx1250) — matrix operand order repaired and checked by reading, not compiled, not run on hardware
//
#include <hip/hip_runtime.h>
#include <math.h>

typedef __bf16 bf16;
typedef __attribute__((ext_vector_type(16))) __bf16 v16bf;
typedef __attribute__((ext_vector_type(8)))  __bf16 v8bf;
typedef __attribute__((ext_vector_type(8)))  float   v8f;
typedef __attribute__((ext_vector_type(4)))  int     v4i;

#define AS1 __attribute__((address_space(1)))
#define AS3 __attribute__((address_space(3)))

#if defined(__has_builtin)
#if __has_builtin(__builtin_amdgcn_global_load_async_to_lds_b128)
#define HAVE_ASYNC_LDS 1
#endif
#endif
#ifndef HAVE_ASYNC_LDS
#define HAVE_ASYNC_LDS 0
#endif

#define R_     512
#define C_     256
#define NC_    80
#define KNMS_  1000
#define NALL_  (R_*NC_)      // 40960
#define IMTOP_ 100
#define KCONV_ 2304          // 9*256
#define KFC0_  12544         // 256*49
#define BBOX_CLIP_ 4.135166556742356f

// ---------------- helpers ----------------
__device__ inline bf16 f2bf(float f) {
  union { float f; unsigned u; } a; a.f = f;
  unsigned r = (a.u + 0x7fffu + ((a.u >> 16) & 1u)) >> 16;
  union { unsigned short s; bf16 b; } o; o.s = (unsigned short)r;
  return o.b;
}

__device__ inline v8f wmma_bf16(v16bf a, v16bf b, v8f c) {
  // D = A(16x32) * B(32x16) + C, f32 accumulate
  return __builtin_amdgcn_wmma_f32_16x16x32_bf16(false, a, false, b, (short)0, c, false, false);
}

// A fragment: 16x32 bf16, lane m = lane&15, K(e) = e + 8h + (e>=8 ? 8 : 0)
__device__ inline v16bf frag_a(const bf16* T, int row, int h) {
  union { v16bf v; v8bf c[2]; } u;
  u.c[0] = *(const v8bf*)(T + row * 32 + 8 * h);
  u.c[1] = *(const v8bf*)(T + row * 32 + 16 + 8 * h);
  return u.v;
}
// B fragment: 32x16 bf16, lane n = lane&15, K(e) = e + 8h + (e>=8 ? 8 : 0)
__device__ inline v16bf frag_b(const bf16* T, int row, int h) {
  union { v16bf v; v8bf c[2]; } u;
  u.c[0] = *(const v8bf*)(T + row * 32 + 8 * h);
  u.c[1] = *(const v8bf*)(T + row * 32 + 16 + 8 * h);
  return u.v;
}

__device__ inline v8bf zero_v8bf() {
  union { v8bf v; unsigned long long q[2]; } z; z.q[0] = 0; z.q[1] = 0; return z.v;
}

// 16B global -> LDS tile staging: async on CDNA5 (GLOBAL_LOAD_ASYNC_TO_LDS_B128,
// tracked by ASYNCcnt), fallback to load+ds_store otherwise.
__device__ inline void stage_b128(const bf16* g, bf16* l) {
#if HAVE_ASYNC_LDS
  __builtin_amdgcn_global_load_async_to_lds_b128(
      (AS1 v4i*)(AS1 void*)g, (AS3 v4i*)(AS3 void*)l, 0, 0);
#else
  *(v8bf*)l = *(const v8bf*)g;
#endif
}

__device__ inline void stage_fence() {
#if HAVE_ASYNC_LDS
#if __has_builtin(__builtin_amdgcn_s_wait_asynccnt)
  __builtin_amdgcn_s_wait_asynccnt(0);
#else
  asm volatile("s_wait_asynccnt 0x0" ::: "memory");
#endif
#endif
}

// ---------------- weight packing ----------------
__global__ void k_f32_to_bf16(const float* __restrict__ s, bf16* __restrict__ d, int n) {
  int i = blockIdx.x * blockDim.x + threadIdx.x;
  if (i < n) d[i] = f2bf(s[i]);
}

// conv w (O,I,3,3) -> [oc][(kh*3+kw)*256+ic] bf16
__global__ void k_pack_conv(const float* __restrict__ w, bf16* __restrict__ d) {
  int i = blockIdx.x * blockDim.x + threadIdx.x;
  if (i >= 256 * KCONV_) return;
  int oc = i / KCONV_, k = i % KCONV_;
  int kh = k / 768, kw = (k / 256) % 3, ic = k & 255;
  d[i] = f2bf(w[((oc * 256 + ic) * 3 + kh) * 3 + kw]);
}

// ---------------- RoI align (writes zero-padded 9x9 channel-last) ----------------
__global__ __launch_bounds__(256)
void k_roi_align(const float* __restrict__ prop, const int* __restrict__ imidx,
                 const float* __restrict__ f0, const float* __restrict__ f1,
                 const float* __restrict__ f2, const float* __restrict__ f3,
                 bf16* __restrict__ pad1) {
  int r = blockIdx.x;
  int c = threadIdx.x;
  float x1 = prop[r * 4 + 0], y1 = prop[r * 4 + 1];
  float x2 = prop[r * 4 + 2], y2 = prop[r * 4 + 3];
  float area = (x2 - x1) * (y2 - y1);
  float kf = floorf(4.0f + log2f(sqrtf(fmaxf(area, 0.f)) / 224.0f + 1e-8f));
  int lvl = (int)(fminf(fmaxf(kf, 2.0f), 5.0f)) - 2;
  const float* fm; int H; float stride;
  if (lvl == 0)      { fm = f0; H = 200; stride = 4.f;  }
  else if (lvl == 1) { fm = f1; H = 100; stride = 8.f;  }
  else if (lvl == 2) { fm = f2; H = 50;  stride = 16.f; }
  else               { fm = f3; H = 25;  stride = 32.f; }
  int W = H;
  int im = imidx[r];
  const float* base = fm + ((size_t)im * C_ + c) * (size_t)(H * W);
  float b0 = x1 / stride, b1 = y1 / stride, b2 = x2 / stride, b3 = y2 / stride;
  float rw = fmaxf(b2 - b0, 1.f), rh = fmaxf(b3 - b1, 1.f);

  for (int py = 0; py < 9; ++py) {
    for (int px = 0; px < 9; ++px) {
      size_t o = ((size_t)r * 81 + py * 9 + px) * C_ + c;
      if (py == 0 || py == 8 || px == 0 || px == 8) { pad1[o] = f2bf(0.f); continue; }
      int i = py - 1, j = px - 1;
      float acc = 0.f;
      for (int sy = 0; sy < 2; ++sy) {
        for (int sx = 0; sx < 2; ++sx) {
          float ty = ((float)(2 * i + sy) + 0.5f) / 14.f;
          float tx = ((float)(2 * j + sx) + 0.5f) / 14.f;
          float ys = fminf(fmaxf(b1 + rh * ty, 0.f), (float)(H - 1));
          float xs = fminf(fmaxf(b0 + rw * tx, 0.f), (float)(W - 1));
          int y0 = (int)floorf(ys), x0 = (int)floorf(xs);
          int y1i = min(y0 + 1, H - 1), x1i = min(x0 + 1, W - 1);
          float ly = ys - (float)y0, lx = xs - (float)x0;
          float v00 = base[y0 * W + x0],  v01 = base[y0 * W + x1i];
          float v10 = base[y1i * W + x0], v11 = base[y1i * W + x1i];
          acc += (1.f - ly) * ((1.f - lx) * v00 + lx * v01)
               +        ly  * ((1.f - lx) * v10 + lx * v11);
        }
      }
      pad1[o] = f2bf(acc * 0.25f);
    }
  }
}

// ---------------- per-RoI implicit-GEMM 3x3 conv via WMMA ----------------
// A = Wpacked[256][2304], B = im2col(pad)[64][32] per K-slab, C = 256 x 49
// mode 0: write padded channel-last output (for next conv); mode 1: write [r][oc*49+p] bf16
__global__ __launch_bounds__(256)
void k_conv(const bf16* __restrict__ pin, const bf16* __restrict__ Wt,
            const float* __restrict__ bias,
            bf16* __restrict__ outPad, bf16* __restrict__ outFc, int mode) {
  __shared__ __align__(16) bf16 At[256 * 32];
  __shared__ __align__(16) bf16 Bt[64 * 32];
  int r = blockIdx.x;
  int t = threadIdx.x;
  int lane = t & 31, wv = t >> 5;
  int h = (lane >> 4) & 1, l16 = lane & 15;
  v8f zero = {0.f, 0.f, 0.f, 0.f, 0.f, 0.f, 0.f, 0.f};
  v8f acc[2][4];
  for (int a = 0; a < 2; ++a)
    for (int b = 0; b < 4; ++b) acc[a][b] = zero;
  const bf16* pr = pin + (size_t)r * 81 * C_;

  for (int k0 = 0; k0 < KCONV_; k0 += 32) {
    // stage A: 256x32 bf16 (1024 x 16B chunks, async global->LDS)
    for (int i = 0; i < 4; ++i) {
      int cid = t + i * 256;
      int oc = cid >> 2, c4 = cid & 3;
      stage_b128(Wt + (size_t)oc * KCONV_ + k0 + c4 * 8, At + oc * 32 + c4 * 8);
    }
    // stage B: im2col 64x32 (one 16B chunk per thread)
    {
      int p = t >> 2, c4 = t & 3, kk = c4 * 8;
      int k = k0 + kk;
      int kh = k / 768, kw = (k / 256) % 3, ic = k & 255;
      if (p < 49) {
        int py = p / 7 + kh, px = p % 7 + kw;
        stage_b128(pr + (py * 9 + px) * C_ + ic, Bt + p * 32 + kk);
      } else {
        *(v8bf*)(Bt + p * 32 + kk) = zero_v8bf();
      }
    }
    stage_fence();
    __syncthreads();
    if (k0 + 32 < KCONV_)
      __builtin_prefetch(Wt + (size_t)(wv * 32) * KCONV_ + k0 + 32, 0, 1);
    v16bf a0 = frag_a(At, wv * 32 + l16, h);
    v16bf a1 = frag_a(At, wv * 32 + 16 + l16, h);
    for (int nt = 0; nt < 4; ++nt) {
      v16bf bfr = frag_b(Bt, nt * 16 + l16, h);
      acc[0][nt] = wmma_bf16(a0, bfr, acc[0][nt]);
      acc[1][nt] = wmma_bf16(a1, bfr, acc[1][nt]);
    }
    __syncthreads();
  }

  // epilogue: bias + ReLU, scatter
  for (int mt = 0; mt < 2; ++mt) {
    for (int nt = 0; nt < 4; ++nt) {
      for (int q = 0; q < 8; ++q) {
        int oc = wv * 32 + mt * 16 + q + 8 * h;
        int p = nt * 16 + l16;
        if (p >= 49) continue;
        float v = acc[mt][nt][q] + bias[oc];
        v = fmaxf(v, 0.f);
        if (mode == 0) {
          int py = p / 7 + 1, px = p % 7 + 1;
          outPad[((size_t)r * 81 + py * 9 + px) * C_ + oc] = f2bf(v);
        } else {
          outFc[(size_t)r * KFC0_ + oc * 49 + p] = f2bf(v);
        }
      }
    }
  }
  if (mode == 0) { // zero the 1-wide border, one channel per thread
    for (int pos = 0; pos < 81; ++pos) {
      int py = pos / 9, px = pos % 9;
      if (py == 0 || py == 8 || px == 0 || px == 8)
        outPad[((size_t)r * 81 + pos) * C_ + t] = f2bf(0.f);
    }
  }
}

// ---------------- generic NT GEMM: out[m][n] = act(sum_k A[m][k]*B[n][k] + bias[n]) ----------------
__global__ __launch_bounds__(128)
void k_gemm_nt(const bf16* __restrict__ A, const bf16* __restrict__ B,
               const float* __restrict__ bias,
               bf16* __restrict__ outb, float* __restrict__ outf,
               int M, int N, int K, int relu) {
  __shared__ __align__(16) bf16 At[64 * 32];
  __shared__ __align__(16) bf16 Bt[64 * 32];
  int bx = blockIdx.x, by = blockIdx.y;
  int t = threadIdx.x;
  int lane = t & 31, wv = t >> 5;
  int h = (lane >> 4) & 1, l16 = lane & 15;
  v8f zero = {0.f, 0.f, 0.f, 0.f, 0.f, 0.f, 0.f, 0.f};
  v8f acc[4];
  for (int n = 0; n < 4; ++n) acc[n] = zero;

  for (int k0 = 0; k0 < K; k0 += 32) {
    for (int i = 0; i < 2; ++i) {
      int cid = t + i * 128;
      int row = cid >> 2, c4 = cid & 3;
      int gm = by * 64 + row;
      if (gm < M)
        stage_b128(A + (size_t)gm * K + k0 + c4 * 8, At + row * 32 + c4 * 8);
      else
        *(v8bf*)(At + row * 32 + c4 * 8) = zero_v8bf();
      int gn = bx * 64 + row;
      if (gn < N)
        stage_b128(B + (size_t)gn * K + k0 + c4 * 8, Bt + row * 32 + c4 * 8);
      else
        *(v8bf*)(Bt + row * 32 + c4 * 8) = zero_v8bf();
    }
    stage_fence();
    __syncthreads();
    if (k0 + 32 < K) {
      __builtin_prefetch(A + (size_t)(by * 64 + wv * 16) * K + k0 + 32, 0, 1);
      __builtin_prefetch(B + (size_t)(bx * 64) * K + k0 + 32, 0, 1);
    }
    v16bf af = frag_a(At, wv * 16 + l16, h);
    for (int nt = 0; nt < 4; ++nt) {
      v16bf bfr = frag_b(Bt, nt * 16 + l16, h);
      acc[nt] = wmma_bf16(af, bfr, acc[nt]);
    }
    __syncthreads();
  }

  for (int nt = 0; nt < 4; ++nt) {
    for (int q = 0; q < 8; ++q) {
      int gm = by * 64 + wv * 16 + q + 8 * h;
      int gn = bx * 64 + nt * 16 + l16;
      if (gm >= M || gn >= N) continue;
      float v = acc[nt][q] + (bias ? bias[gn] : 0.f);
      if (relu) v = fmaxf(v, 0.f);
      if (outb) outb[(size_t)gm * N + gn] = f2bf(v);
      if (outf) outf[(size_t)gm * N + gn] = v;
    }
  }
}

// ---------------- softmax over 81 classes, keep [1:] ----------------
__global__ void k_softmax(const float* __restrict__ logits, float* __restrict__ scr) {
  int r = blockIdx.x * blockDim.x + threadIdx.x;
  if (r >= R_) return;
  const float* l = logits + (size_t)r * 81;
  float mx = l[0];
  for (int i = 1; i < 81; ++i) mx = fmaxf(mx, l[i]);
  float s = 0.f;
  for (int i = 0; i < 81; ++i) s += __expf(l[i] - mx);
  float inv = 1.f / s;
  for (int j = 1; j < 81; ++j) scr[(size_t)r * NC_ + (j - 1)] = __expf(l[j] - mx) * inv;
}

// ---------------- box decode + validity + class/image offsets ----------------
__global__ void k_decode(const float* __restrict__ prop, const int* __restrict__ imidx,
                         const float* __restrict__ imsz, const float* __restrict__ reg,
                         const float* __restrict__ scr,
                         float* __restrict__ bflat, float* __restrict__ obox,
                         float* __restrict__ sflat) {
  int idx = blockIdx.x * blockDim.x + threadIdx.x;
  if (idx >= NALL_) return;
  int r = idx / NC_, j = idx % NC_;
  int cid = j + 1;
  int im = imidx[r];
  float p0 = prop[r * 4 + 0], p1 = prop[r * 4 + 1];
  float p2 = prop[r * 4 + 2], p3 = prop[r * 4 + 3];
  float pw = p2 - p0, ph = p3 - p1;
  float px = p0 + pw * 0.5f, py = p1 + ph * 0.5f;
  const float* rg = reg + (size_t)r * (NC_ * 4) + j * 4;
  float cx = px + rg[0] * 0.1f * pw;
  float cy = py + rg[1] * 0.1f * ph;
  float bw = pw * expf(fminf(rg[2] * 0.2f, BBOX_CLIP_));
  float bh = ph * expf(fminf(rg[3] * 0.2f, BBOX_CLIP_));
  float hwH = imsz[im * 2 + 0], hwW = imsz[im * 2 + 1];
  float x0 = fminf(fmaxf(cx - bw * 0.5f, 0.f), hwW);
  float y0 = fminf(fmaxf(cy - bh * 0.5f, 0.f), hwH);
  float x1 = fminf(fmaxf(cx + bw * 0.5f, 0.f), hwW);
  float y1 = fminf(fmaxf(cy + bh * 0.5f, 0.f), hwH);
  float ww = x1 - x0, hh = y1 - y0;
  float s = scr[idx];
  bool valid = (s > 0.05f) && (ww >= 1.0f) && (hh >= 1.0f);
  sflat[idx] = valid ? s : 0.f;
  bflat[idx * 4 + 0] = x0; bflat[idx * 4 + 1] = y0;
  bflat[idx * 4 + 2] = x1; bflat[idx * 4 + 3] = y1;
  float off = (float)(im * (NC_ + 1) + cid) * 4096.0f;
  obox[idx * 4 + 0] = x0 + off; obox[idx * 4 + 1] = y0 + off;
  obox[idx * 4 + 2] = x1 + off; obox[idx * 4 + 3] = y1 + off;
}

// ---------------- top-1000 by stable rank (matches lax.top_k) ----------------
__global__ __launch_bounds__(256)
void k_topk(const float* __restrict__ sflat, const float* __restrict__ bflat,
            const float* __restrict__ obox, const int* __restrict__ imidx,
            float* __restrict__ topv, int* __restrict__ topi,
            float* __restrict__ kb, float* __restrict__ kbox,
            int* __restrict__ kc, int* __restrict__ ki) {
  __shared__ float sc[256];
  int i = blockIdx.x * 256 + threadIdx.x; // grid covers NALL_ exactly
  float si = sflat[i];
  int rank = 0;
  for (int j0 = 0; j0 < NALL_; j0 += 256) {
    sc[threadIdx.x] = sflat[j0 + threadIdx.x];
    __syncthreads();
    for (int jj = 0; jj < 256; ++jj) {
      int j = j0 + jj;
      float sj = sc[jj];
      rank += (int)((sj > si) || (sj == si && j < i));
    }
    __syncthreads();
  }
  if (rank < KNMS_) {
    topv[rank] = si;
    topi[rank] = i;
    for (int q = 0; q < 4; ++q) {
      kb[rank * 4 + q] = obox[i * 4 + q];
      kbox[rank * 4 + q] = bflat[i * 4 + q];
    }
    kc[rank] = (i % NC_) + 1;
    ki[rank] = imidx[i / NC_];
  }
}

// ---------------- sequential-suppression NMS (matches lax.scan) ----------------
__global__ __launch_bounds__(1024)
void k_nms(const float* __restrict__ kb, const float* __restrict__ topv,
           float* __restrict__ fs) {
  __shared__ float bx0[KNMS_], by0[KNMS_], bx1[KNMS_], by1[KNMS_], ar[KNMS_], tv[KNMS_];
  __shared__ int sup[KNMS_];
  int t = threadIdx.x;
  for (int j = t; j < KNMS_; j += 1024) {
    float a0 = kb[j * 4 + 0], a1 = kb[j * 4 + 1];
    float a2 = kb[j * 4 + 2], a3 = kb[j * 4 + 3];
    bx0[j] = a0; by0[j] = a1; bx1[j] = a2; by1[j] = a3;
    ar[j] = (a2 - a0) * (a3 - a1);
    tv[j] = topv[j];
    sup[j] = 0;
  }
  __syncthreads();
  for (int i = 0; i < KNMS_; ++i) {
    bool keep = (sup[i] == 0) && (tv[i] > 0.f);
    if (keep && t < KNMS_ && t > i) {
      float ltx = fmaxf(bx0[i], bx0[t]);
      float lty = fmaxf(by0[i], by0[t]);
      float rbx = fminf(bx1[i], bx1[t]);
      float rby = fminf(by1[i], by1[t]);
      float wi = fmaxf(rbx - ltx, 0.f), hi = fmaxf(rby - lty, 0.f);
      float inter = wi * hi;
      float iou = inter / (ar[i] + ar[t] - inter + 1e-9f);
      if (iou > 0.5f) sup[t] = 1;
    }
    __syncthreads();
  }
  for (int j = t; j < KNMS_; j += 1024)
    fs[j] = ((sup[j] == 0) && (tv[j] > 0.f)) ? tv[j] : 0.f;
}

// ---------------- per-image top-100 + write outputs ----------------
__global__ __launch_bounds__(256)
void k_final(const float* __restrict__ fs, const float* __restrict__ kbox,
             const int* __restrict__ kc, const int* __restrict__ ki,
             float* __restrict__ out) {
  __shared__ float si[KNMS_];
  int img = blockIdx.x;
  int t = threadIdx.x;
  for (int j = t; j < KNMS_; j += 256) si[j] = (ki[j] == img) ? fs[j] : 0.f;
  __syncthreads();
  for (int j = t; j < KNMS_; j += 256) {
    float sj = si[j];
    int rank = 0;
    for (int q = 0; q < KNMS_; ++q)
      rank += (int)((si[q] > sj) || (si[q] == sj && q < j));
    if (rank < IMTOP_) {
      int o = img * IMTOP_ + rank;
      out[o * 4 + 0] = kbox[j * 4 + 0];
      out[o * 4 + 1] = kbox[j * 4 + 1];
      out[o * 4 + 2] = kbox[j * 4 + 2];
      out[o * 4 + 3] = kbox[j * 4 + 3];
      out[800 + o] = sj;
      out[1000 + o] = (float)kc[j];
    }
  }
}

// ---------------- host launch ----------------
extern "C" void kernel_launch(void* const* d_in, const int* in_sizes, int n_in,
                              void* d_out, int out_size, void* d_ws, size_t ws_size,
                              hipStream_t stream) {
  (void)in_sizes; (void)n_in; (void)out_size; (void)ws_size;
  const float* proposals = (const float*)d_in[0];
  const int*   imidx     = (const int*)d_in[1];
  const float* fmap0     = (const float*)d_in[2];
  const float* fmap1     = (const float*)d_in[3];
  const float* fmap2     = (const float*)d_in[4];
  const float* fmap3     = (const float*)d_in[5];
  const float* imsizes   = (const float*)d_in[6];
  const float* conv0_w   = (const float*)d_in[7];
  const float* conv0_b   = (const float*)d_in[8];
  const float* conv1_w   = (const float*)d_in[9];
  const float* conv1_b   = (const float*)d_in[10];
  const float* fc0_w     = (const float*)d_in[11];
  const float* fc0_b     = (const float*)d_in[12];
  const float* fc1_w     = (const float*)d_in[13];
  const float* fc1_b     = (const float*)d_in[14];
  const float* cls_w     = (const float*)d_in[15];
  const float* cls_b     = (const float*)d_in[16];
  const float* reg_w     = (const float*)d_in[17];
  const float* reg_b     = (const float*)d_in[18];

  unsigned char* wp = (unsigned char*)d_ws;
  auto alloc = [&](size_t bytes) -> void* {
    void* r = (void*)wp;
    wp += (bytes + 255) & ~(size_t)255;
    return r;
  };

  bf16*  pad1  = (bf16*) alloc((size_t)R_ * 81 * C_ * 2);
  bf16*  pad2  = (bf16*) alloc((size_t)R_ * 81 * C_ * 2);
  bf16*  fcX   = (bf16*) alloc((size_t)R_ * KFC0_ * 2);
  bf16*  fc0o  = (bf16*) alloc((size_t)R_ * 1024 * 2);
  bf16*  fc1o  = (bf16*) alloc((size_t)R_ * 1024 * 2);
  float* clsL  = (float*)alloc((size_t)R_ * 81 * 4);
  float* regO  = (float*)alloc((size_t)R_ * 320 * 4);
  float* scr   = (float*)alloc((size_t)R_ * NC_ * 4);
  bf16*  w0c   = (bf16*) alloc((size_t)256 * KCONV_ * 2);
  bf16*  w1c   = (bf16*) alloc((size_t)256 * KCONV_ * 2);
  bf16*  wf0   = (bf16*) alloc((size_t)1024 * KFC0_ * 2);
  bf16*  wf1   = (bf16*) alloc((size_t)1024 * 1024 * 2);
  bf16*  wcl   = (bf16*) alloc((size_t)81 * 1024 * 2);
  bf16*  wrg   = (bf16*) alloc((size_t)320 * 1024 * 2);
  float* bflat = (float*)alloc((size_t)NALL_ * 4 * 4);
  float* obox  = (float*)alloc((size_t)NALL_ * 4 * 4);
  float* sflat = (float*)alloc((size_t)NALL_ * 4);
  float* topv  = (float*)alloc((size_t)KNMS_ * 4);
  int*   topi  = (int*)  alloc((size_t)KNMS_ * 4);
  float* kb    = (float*)alloc((size_t)KNMS_ * 4 * 4);
  float* kbox  = (float*)alloc((size_t)KNMS_ * 4 * 4);
  int*   kc    = (int*)  alloc((size_t)KNMS_ * 4);
  int*   ki    = (int*)  alloc((size_t)KNMS_ * 4);
  float* fs    = (float*)alloc((size_t)KNMS_ * 4);

  // 1) weight repack / convert to bf16
  k_pack_conv<<<(256 * KCONV_ + 255) / 256, 256, 0, stream>>>(conv0_w, w0c);
  k_pack_conv<<<(256 * KCONV_ + 255) / 256, 256, 0, stream>>>(conv1_w, w1c);
  k_f32_to_bf16<<<(1024 * KFC0_ + 255) / 256, 256, 0, stream>>>(fc0_w, wf0, 1024 * KFC0_);
  k_f32_to_bf16<<<(1024 * 1024 + 255) / 256, 256, 0, stream>>>(fc1_w, wf1, 1024 * 1024);
  k_f32_to_bf16<<<(81 * 1024 + 255) / 256, 256, 0, stream>>>(cls_w, wcl, 81 * 1024);
  k_f32_to_bf16<<<(320 * 1024 + 255) / 256, 256, 0, stream>>>(reg_w, wrg, 320 * 1024);

  // 2) RoI align -> padded channel-last tiles
  k_roi_align<<<R_, 256, 0, stream>>>(proposals, imidx, fmap0, fmap1, fmap2, fmap3, pad1);

  // 3) conv0 (WMMA implicit GEMM) -> padded; conv1 -> flattened fc layout
  k_conv<<<R_, 256, 0, stream>>>(pad1, w0c, conv0_b, pad2, nullptr, 0);
  k_conv<<<R_, 256, 0, stream>>>(pad2, w1c, conv1_b, nullptr, fcX, 1);

  // 4) FC stack + heads (WMMA NT GEMMs, fused bias/ReLU)
  dim3 gf(1024 / 64, R_ / 64);
  k_gemm_nt<<<gf, 128, 0, stream>>>(fcX, wf0, fc0_b, fc0o, nullptr, R_, 1024, KFC0_, 1);
  k_gemm_nt<<<gf, 128, 0, stream>>>(fc0o, wf1, fc1_b, fc1o, nullptr, R_, 1024, 1024, 1);
  dim3 gc((81 + 63) / 64, R_ / 64);
  k_gemm_nt<<<gc, 128, 0, stream>>>(fc1o, wcl, cls_b, nullptr, clsL, R_, 81, 1024, 0);
  dim3 gr((320 + 63) / 64, R_ / 64);
  k_gemm_nt<<<gr, 128, 0, stream>>>(fc1o, wrg, reg_b, nullptr, regO, R_, 320, 1024, 0);

  // 5) tail: softmax, decode, top-k, NMS, per-image top-100
  k_softmax<<<(R_ + 255) / 256, 256, 0, stream>>>(clsL, scr);
  k_decode<<<NALL_ / 256, 256, 0, stream>>>(proposals, imidx, imsizes, regO, scr,
                                            bflat, obox, sflat);
  k_topk<<<NALL_ / 256, 256, 0, stream>>>(sflat, bflat, obox, imidx,
                                          topv, topi, kb, kbox, kc, ki);
  k_nms<<<1, 1024, 0, stream>>>(kb, topv, fs);
  k_final<<<2, 256, 0, stream>>>(fs, kbox, kc, ki, (float*)d_out);
}
